// CausalGraphEncoder_2637109919793
// MI455X (gfx1250) — matrix operand order repaired and checked by reading, not compiled, not run on hardware
//
#include <hip/hip_runtime.h>
#include <hip/hip_bf16.h>

typedef __attribute__((ext_vector_type(16))) __bf16 v16bf;
typedef __attribute__((ext_vector_type(8)))  float  v8f;
typedef __attribute__((ext_vector_type(4)))  unsigned int u32x4;
typedef __attribute__((ext_vector_type(8)))  unsigned int u32x8;

#define D_MODEL 512
#define SEQ     4096

// Native float->bf16 conversions (v_cvt_pk_bf16_f32 on gfx1250).
__device__ __forceinline__ unsigned short f2bf(float f) {
    union { __bf16 b; unsigned short u; } cv;
    cv.b = (__bf16)f;
    return cv.u;
}
__device__ __forceinline__ unsigned int pack2bf(float a, float b) {
    union { __bf16 b[2]; unsigned int u; } cv;
    cv.b[0] = (__bf16)a;
    cv.b[1] = (__bf16)b;
    return cv.u;
}

union Frag16 { unsigned int u[8]; unsigned short s[16]; v16bf v; };

// ---------------------------------------------------------------------------
// TDM: DMA one contiguous 32x512 bf16 tile (32 KB) from global to LDS.
// D# descriptor per CDNA5 ISA ch.8: 1-D tile, 16384 elements of 2 bytes.
// Issued by one wave; completion tracked with TENSORcnt.
// ---------------------------------------------------------------------------
__device__ __forceinline__ void tdm_load_tile(const unsigned short* gsrc,
                                              unsigned short* ldst) {
    unsigned long long ga = (unsigned long long)(size_t)gsrc;
    unsigned           la = (unsigned)(size_t)ldst;          // LDS byte offset
    u32x4 g0;
    g0[0] = 1u;                                              // count=1, user D#
    g0[1] = la;                                              // lds_addr
    g0[2] = (unsigned)ga;                                    // global_addr[31:0]
    g0[3] = (unsigned)((ga >> 32) & 0x01FFFFFFu)             // global_addr[56:32]
          | 0x80000000u;                                     // type=2 ("image")
    u32x8 g1;
    g1[0] = 0x00010000u;     // workgroup_mask=0, data_size=1 (2 bytes)
    g1[1] = 0x40000000u;     // tensor_dim0 = 16384  (bits 79:48)
    g1[2] = 0x00010000u;     // tensor_dim1 = 1      (bits 111:80)
    g1[3] = 0x40000000u;     // tile_dim0   = 16384  (bits 127:112)
    g1[4] = 0x00000001u;     // tile_dim1 = 1, tile_dim2 = 0
    g1[5] = 16384u;          // tensor_dim0_stride lo
    g1[6] = 0u;              // stride0 hi, tensor_dim1_stride lo16
    g1[7] = 0u;              // tensor_dim1_stride hi
    u32x4 g2 = {0u, 0u, 0u, 0u};
    u32x4 g3 = {0u, 0u, 0u, 0u};
    asm volatile("tensor_load_to_lds %0, %1, %2, %3"
                 :: "s"(g0), "s"(g1), "s"(g2), "s"(g3)
                 : "memory");
}

// ---------------------------------------------------------------------------
// Kernel 1: node = relu(x @ W^T + b), stored as bf16 to workspace.
// One wave per 16x16 output tile; K=512 contracted in 16 WMMA steps.
// ---------------------------------------------------------------------------
__global__ __launch_bounds__(256)
void node_proj_kernel(const float* __restrict__ x, const float* __restrict__ W,
                      const float* __restrict__ bias, unsigned short* __restrict__ node) {
    const int wave = threadIdx.x >> 5;
    const int lane = threadIdx.x & 31;
    const int h    = lane >> 4;
    const int ln   = lane & 15;

    const int tile  = blockIdx.x * 8 + wave;   // 32768 tiles total
    const int mtile = tile >> 5;               // 1024 tiles of M
    const int etile = tile & 31;               // 32 tiles of E

    const int arow = mtile * 16 + ln;          // A-matrix row held by this lane
    const int ecol = etile * 16 + ln;          // B-matrix column held by this lane

    v8f acc = {};
    #pragma unroll
    for (int c = 0; c < 16; ++c) {
        const int kb = c * 32;
        Frag16 a, b;
        #pragma unroll
        for (int j = 0; j < 8; ++j) {
            // A layout: dword j holds K = 16h+2j, +1
            int dl = 16 * h + 2 * j;
            int d  = kb + dl;
            a.u[j] = pack2bf(x[arow * D_MODEL + d], x[arow * D_MODEL + d + 1]);
            // B layout: dword j holds K = 16h+2j, +1 ; B[d,e] = W[e,d]
            int dB = kb + 16 * h + 2 * j;
            b.u[j] = pack2bf(W[ecol * D_MODEL + dB], W[ecol * D_MODEL + dB + 1]);
        }
        acc = __builtin_amdgcn_wmma_f32_16x16x32_bf16(false, a.v, false, b.v,
                                                      (short)0, acc, false, false);
    }

    const float bv = bias[ecol];
    #pragma unroll
    for (int r = 0; r < 8; ++r) {
        int m   = mtile * 16 + r + 8 * h;   // C layout: VGPR r -> M = r + 8*half
        float v = acc[r] + bv;
        v = v > 0.f ? v : 0.f;              // relu
        node[m * D_MODEL + ecol] = f2bf(v);
    }
}

// ---------------------------------------------------------------------------
// Kernel 2: fused flash attention  out = softmax(node @ node^T) @ node
// Block = 128 threads (4 waves), one 16-row query tile per block.
// K/V tiles are double-buffered in LDS and staged by the Tensor Data Mover
// (wave 0 issues tensor_load_to_lds for tile kt+1 while all waves compute
// on tile kt). Online softmax; scores never touch HBM.
// ---------------------------------------------------------------------------
__global__ __launch_bounds__(128)
void flash_attn_kernel(const unsigned short* __restrict__ node, float* __restrict__ out) {
    __shared__ __align__(16) unsigned short KtBuf[2][32 * D_MODEL]; // 64 KB ping-pong
    __shared__ __align__(16) unsigned short Qt[16 * D_MODEL];       // 16 KB
    __shared__ __align__(16) unsigned short Pt[4][16 * 32];         // per-wave P, 4 KB
    __shared__ float mstat[4][16];
    __shared__ float lstat[4][16];

    const int tid  = threadIdx.x;
    const int wave = tid >> 5;
    const int lane = tid & 31;
    const int h    = lane >> 4;
    const int ln   = lane & 15;

    const int  batch = blockIdx.x >> 8;           // 256 query tiles per batch
    const int  qtile = blockIdx.x & 255;
    const long qrow0 = (long)batch * SEQ + (long)qtile * 16;
    const long krow0 = (long)batch * SEQ;

    // Kick off TDM for key tile 0 into buffer 0 (wave 0 only).
    if (wave == 0)
        tdm_load_tile(node + krow0 * D_MODEL, &KtBuf[0][0]);

    // Stage Q tile (16x512 bf16 = 1024 uint4) cooperatively.
    {
        const uint4* src = (const uint4*)(node + qrow0 * D_MODEL);
        uint4*       dst = (uint4*)Qt;
        #pragma unroll
        for (int i = 0; i < 8; ++i) dst[tid + 128 * i] = src[tid + 128 * i];
    }
    if (lane < 16) { mstat[wave][lane] = -1e30f; lstat[wave][lane] = 0.f; }

    v8f acc[8];
    #pragma unroll
    for (int nt = 0; nt < 8; ++nt) acc[nt] = (v8f){};

    const unsigned int* Qt32 = (const unsigned int*)Qt;
    const unsigned int* Pt32 = (const unsigned int*)Pt[wave];

    for (int kt = 0; kt < SEQ / 32; ++kt) {
        const int buf = kt & 1;
        if (wave == 0) __builtin_amdgcn_s_wait_tensorcnt(0);  // tile kt landed
        __syncthreads();  // tile kt visible to all; all waves done with buf^1
        if (wave == 0 && kt + 1 < SEQ / 32)                   // prefetch kt+1
            tdm_load_tile(node + (krow0 + (long)(kt + 1) * 32) * D_MODEL,
                          &KtBuf[buf ^ 1][0]);

        const unsigned short* Kt   = KtBuf[buf];
        const unsigned int*   Kt32 = (const unsigned int*)Kt;

        // ---- scores: S[16x32] = Q(16x512) @ K^T, two 16x16 WMMA accumulators
        v8f s0 = {}, s1 = {};
        #pragma unroll
        for (int c = 0; c < 16; ++c) {
            Frag16 qa, b0, b1;
            #pragma unroll
            for (int j = 0; j < 8; ++j) {
                int dl = 16 * h + 2 * j;
                qa.u[j] = Qt32[(ln * D_MODEL + c * 32 + dl) >> 1];
                int dB  = c * 32 + 16 * h + 2 * j;          // B contraction = d
                b0.u[j] = Kt32[((ln)      * D_MODEL + dB) >> 1];  // keys 0..15
                b1.u[j] = Kt32[((ln + 16) * D_MODEL + dB) >> 1];  // keys 16..31
            }
            s0 = __builtin_amdgcn_wmma_f32_16x16x32_bf16(false, qa.v, false, b0.v,
                                                         (short)0, s0, false, false);
            s1 = __builtin_amdgcn_wmma_f32_16x16x32_bf16(false, qa.v, false, b1.v,
                                                         (short)0, s1, false, false);
        }

        // ---- online softmax over this 32-key tile
        float corr[8];
        #pragma unroll
        for (int r = 0; r < 8; ++r) {
            const int row = r + 8 * h;                     // C layout row
            float a0 = s0[r], a1 = s1[r];
            float mx = fmaxf(a0, a1);
            mx = fmaxf(mx, __shfl_xor(mx, 1, 32));
            mx = fmaxf(mx, __shfl_xor(mx, 2, 32));
            mx = fmaxf(mx, __shfl_xor(mx, 4, 32));
            mx = fmaxf(mx, __shfl_xor(mx, 8, 32));         // row max within half
            float mold = mstat[wave][row];
            float mnew = fmaxf(mold, mx);
            float cf   = __expf(mold - mnew);
            float p0   = __expf(a0 - mnew);
            float p1   = __expf(a1 - mnew);
            float rs   = p0 + p1;
            rs += __shfl_xor(rs, 1, 32);
            rs += __shfl_xor(rs, 2, 32);
            rs += __shfl_xor(rs, 4, 32);
            rs += __shfl_xor(rs, 8, 32);
            if (ln == 0) {
                mstat[wave][row] = mnew;
                lstat[wave][row] = lstat[wave][row] * cf + rs;
            }
            Pt[wave][row * 32 + ln]      = f2bf(p0);       // P in row-major LDS
            Pt[wave][row * 32 + 16 + ln] = f2bf(p1);
            corr[r] = cf;
        }

        // ---- O = O*corr + P(16x32) @ V(32x128-slice), 8 WMMAs
        Frag16 pa;
        #pragma unroll
        for (int j = 0; j < 8; ++j) {
            int kl = 16 * h + 2 * j;
            pa.u[j] = Pt32[(ln * 32 + kl) >> 1];
        }
        #pragma unroll
        for (int nt = 0; nt < 8; ++nt) {
            #pragma unroll
            for (int r = 0; r < 8; ++r) acc[nt][r] *= corr[r];
            Frag16 vb;
            const int dcol = wave * 128 + nt * 16 + ln;    // B column = output dim
            #pragma unroll
            for (int e = 0; e < 16; ++e)                   // element e <-> key 16h+e
                vb.s[e] = Kt[(16 * h + e) * D_MODEL + dcol];
            acc[nt] = __builtin_amdgcn_wmma_f32_16x16x32_bf16(false, pa.v, false, vb.v,
                                                              (short)0, acc[nt], false, false);
        }
    }

    // ---- finalize: divide by softmax denominator, write f32 output
    #pragma unroll
    for (int nt = 0; nt < 8; ++nt) {
        const int dcol = wave * 128 + nt * 16 + ln;
        #pragma unroll
        for (int r = 0; r < 8; ++r) {
            const int row  = r + 8 * h;
            const float dn = lstat[wave][row];
            out[(qrow0 + row) * D_MODEL + dcol] = acc[nt][r] / dn;
        }
    }
}

// ---------------------------------------------------------------------------
extern "C" void kernel_launch(void* const* d_in, const int* in_sizes, int n_in,
                              void* d_out, int out_size, void* d_ws, size_t ws_size,
                              hipStream_t stream) {
    const float* x = (const float*)d_in[0];   // [4,4096,512] f32
    const float* W = (const float*)d_in[1];   // [512,512]    f32
    const float* b = (const float*)d_in[2];   // [512]        f32
    float* out = (float*)d_out;               // [4,4096,512] f32
    unsigned short* node = (unsigned short*)d_ws;  // bf16 node features, 16.8 MB

    // 16384x512 output tiles of 16x16 -> 32768 wave-tiles, 8 waves/block
    node_proj_kernel<<<4096, 256, 0, stream>>>(x, W, b, node);
    // one block per 16-row query tile: 4 batches * 256 tiles
    flash_attn_kernel<<<1024, 128, 0, stream>>>(node, out);
}
